// TemporalPredictionNetwork_36756330119914
// MI455X (gfx1250) — matrix operand order repaired and checked by reading, not compiled, not run on hardware
//
#include <hip/hip_runtime.h>
#include <hip/hip_bf16.h>
#include <math.h>

// ---------------------------------------------------------------------------
// TemporalPredictionNetwork on MI455X (gfx1250), compile-only tuning.
//
//  * B=1 GEMV chain, 512 sequential steps -> memory/latency bound.
//    Weights converted once to bf16 (64MB, resident in 192MB L2); every
//    matvec runs through v_wmma_f32_16x16x32_bf16 (fp32 accumulate).
//  * One persistent kernel (16 WGs x 512 thr = 256 wave32 waves) with a
//    device-scope sense-reversal barrier between stages (no kernel-launch
//    overhead in the 512-step loop).
//  * wih rows are permuted at conversion time (row' = 4*j + gate) so the
//    LSTM activation fuses into the gate-GEMV epilogue: one fewer global
//    barrier per level and no g4/upd float round-trips.
//  * GEMV inputs except err at l>0 (fp32 inp - pred, converted in-loop) are bf16 mirrors written by the
//    producing epilogue (or once, for `observed`), so B-fragments are two
//    b128 loads instead of 4 float4 loads + 8 v_cvt_pk_bf16_f32.
// Workspace: 64MB bf16 weights + 1MB bf16 observed + ~30KB state.
// ---------------------------------------------------------------------------

typedef __bf16 bf16_t;
typedef __attribute__((ext_vector_type(8)))  __bf16 bf16x8;
typedef __attribute__((ext_vector_type(16))) __bf16 v16bf;
typedef __attribute__((ext_vector_type(8)))  float  v8f;

static constexpr int  L = 4, H = 1024, S = 512;
static constexpr long NHH = 4L * 1024 * 1024;      // L*H*H
static constexpr long NW  = 16L * 1024 * 1024;     // L*4H*H
static constexpr long WB_ELEMS = 4 * NHH + NW;     // bf16 weight elems (64MB)
static constexpr long OBS_ELEMS = (long)S * H;     // bf16 observed mirror
static constexpr long CONV_TOTAL = WB_ELEMS + OBS_ELEMS;

static constexpr int NWG = 16;
static constexpr int BLK = 512;                    // 16 wave32 waves / WG
static constexpr int NWAVES = NWG * BLK / 32;      // 256

// float-scratch layout (indices into fs[])
static constexpr int FS_HS   = 0;        // L*H fp32 LSTM state
static constexpr int FS_PREC = 4096;
static constexpr int FS_FE   = 4097;
static constexpr int FS_FEP  = 4160;     // NWAVES partials
static constexpr int FS_BAR  = 4416;     // 2 x u32 barrier state
static constexpr int FS_BF   = 4480;     // bf16 mirrors start here (as bf16*)
// bf16 mirror layout (indices into bss[])
static constexpr int BS_T1  = 0;         // H
static constexpr int BS_UPD = 1024;      // H
static constexpr int BS_HS  = 2048;      // L*H

__device__ __forceinline__ float gelu_exact(float x) {
  return 0.5f * x * (1.0f + erff(x * 0.7071067811865476f));
}
__device__ __forceinline__ float sigm(float x) { return 1.0f / (1.0f + expf(-x)); }

// Device-wide sense-reversal barrier (16 co-resident WGs).
__device__ __forceinline__ void gbar(unsigned* cnt, unsigned* gen) {
  __threadfence();
  __syncthreads();
  if (threadIdx.x == 0) {
    unsigned g = __hip_atomic_load(gen, __ATOMIC_RELAXED, __HIP_MEMORY_SCOPE_AGENT);
    unsigned a = __hip_atomic_fetch_add(cnt, 1u, __ATOMIC_ACQ_REL, __HIP_MEMORY_SCOPE_AGENT);
    if (a == (unsigned)(gridDim.x - 1)) {
      __hip_atomic_store(cnt, 0u, __ATOMIC_RELAXED, __HIP_MEMORY_SCOPE_AGENT);
      __hip_atomic_store(gen, g + 1u, __ATOMIC_RELEASE, __HIP_MEMORY_SCOPE_AGENT);
    } else {
      while (__hip_atomic_load(gen, __ATOMIC_ACQUIRE, __HIP_MEMORY_SCOPE_AGENT) == g)
        __builtin_amdgcn_s_sleep(1);
    }
  }
  __syncthreads();
  __threadfence();
}

enum XMode { XBF16, XF32SUB };

// y[rows] = W[rows x K](bf16 row-major) * x[K]   via v_wmma_f32_16x16x32_bf16.
// One 16-row tile per wave.  A-frag per ISA 7.12.2 (16-bit A 16x32): lane<16
// loads K [kb..kb+7] and [kb+16..kb+23] as b128 pairs, lane>=16 the rest.
// B-frag = x broadcast into all 16 columns; lanes 0/16 (N==0) own the 8+8
// result rows in the 8 accumulator VGPRs.  Epilogue gets (m0, v8f acc).
template <XMode XM, typename Epi>
__device__ __forceinline__ void gemv_bf16(const bf16_t* __restrict__ W, int rows, int K,
                                          const void* __restrict__ xv,
                                          const float* __restrict__ xs,
                                          int wid, Epi epi) {
  const int lane = threadIdx.x & 31;
  const int half = lane >> 4;
  const int lrow = lane & 15;
  const int ntiles = rows >> 4;
  for (int rt = wid; rt < ntiles; rt += NWAVES) {
    const bf16_t* wrow = W + (size_t)(rt * 16 + lrow) * (size_t)K;
    v8f acc = {0.f, 0.f, 0.f, 0.f, 0.f, 0.f, 0.f, 0.f};
    for (int kb = 0; kb < K; kb += 32) {
      union { v16bf v; bf16x8 h[2]; bf16_t e[16]; } a, b;
      const bf16_t* wp = wrow + kb + half * 8;
      a.h[0] = *(const bf16x8*)(wp);            // b128
      a.h[1] = *(const bf16x8*)(wp + 16);       // b128
      const int kx = kb + half * 8;
      if constexpr (XM == XBF16) {
        const bf16_t* xb = (const bf16_t*)xv;
        b.h[0] = *(const bf16x8*)(xb + kx);     // b128
        b.h[1] = *(const bf16x8*)(xb + kx + 16); // b128
      } else {                                  // fp32 (inp - pred) on the fly
        const float* x = (const float*)xv;
        float fv[16];
        const float4* xp = (const float4*)(x + kx);
        *(float4*)&fv[0]  = xp[0]; *(float4*)&fv[4]  = xp[1];
        *(float4*)&fv[8]  = xp[4]; *(float4*)&fv[12] = xp[5];
        float sv[16];
        const float4* sp = (const float4*)(xs + kx);
        *(float4*)&sv[0]  = sp[0]; *(float4*)&sv[4]  = sp[1];
        *(float4*)&sv[8]  = sp[4]; *(float4*)&sv[12] = sp[5];
#pragma unroll
        for (int i = 0; i < 16; ++i) b.e[i] = (bf16_t)(fv[i] - sv[i]);
      }
      acc = __builtin_amdgcn_wmma_f32_16x16x32_bf16(
          false, a.v, false, b.v, (short)0, acc, false, false);
    }
    if (lrow == 0) epi(rt * 16 + half * 8, acc);
  }
}

// Converts all weights to bf16 (wih with rows permuted to 4*j+gate) and
// mirrors `observed` to bf16.  Runs once per launch; traffic is negligible
// vs. the 512x re-streaming of the weights themselves.
__global__ void convert_weights(const float* __restrict__ pw1, const float* __restrict__ pw2,
                                const float* __restrict__ ew1, const float* __restrict__ ew2,
                                const float* __restrict__ wih, const float* __restrict__ obs,
                                bf16_t* __restrict__ dst) {
  const size_t stride = (size_t)gridDim.x * blockDim.x;
  for (size_t i = (size_t)blockIdx.x * blockDim.x + threadIdx.x;
       i < (size_t)CONV_TOTAL; i += stride) {
    float v;
    if      (i < (size_t)(1 * NHH)) v = pw1[i];
    else if (i < (size_t)(2 * NHH)) v = pw2[i - 1 * NHH];
    else if (i < (size_t)(3 * NHH)) v = ew1[i - 2 * NHH];
    else if (i < (size_t)(4 * NHH)) v = ew2[i - 3 * NHH];
    else if (i < (size_t)(4 * NHH + NW)) {       // wih, gate-interleaved rows
      size_t r  = i - 4 * NHH;
      size_t lvl = r >> 22;                      // / (4*H*H)
      size_t rr  = r & ((1u << 22) - 1);
      int rp = (int)(rr >> 10);                  // permuted row (4*j + gate)
      int k  = (int)(rr & (H - 1));
      int j = rp >> 2, gate = rp & 3;
      v = wih[lvl * 4 * H * H + ((size_t)gate * H + j) * H + k];
    } else {
      v = obs[i - (4 * NHH + NW)];
    }
    dst[i] = (bf16_t)v;
  }
}

__global__ void init_state(float* fs, float* out) {
  const int i = blockIdx.x * blockDim.x + threadIdx.x;
  bf16_t* bss = (bf16_t*)(fs + FS_BF);
  if (i < L * H) { fs[FS_HS + i] = 0.f; bss[BS_HS + i] = (bf16_t)0.f; }
  if (i < H)     out[i] = 0.f;          // level-0 pred == 0 forever
  if (i == 0) {
    fs[FS_FE] = 0.f;
    unsigned* b = (unsigned*)(fs + FS_BAR);
    b[0] = 0u; b[1] = 0u;               // barrier count / generation
  }
}

__global__ __launch_bounds__(BLK)
void tpn_persistent(const float* __restrict__ obs,
                    const float* __restrict__ pb1, const float* __restrict__ pb2,
                    const float* __restrict__ eb1, const float* __restrict__ eb2,
                    const float* __restrict__ qw,  const float* __restrict__ qb,
                    const float* __restrict__ bih, const float* __restrict__ bhh,
                    const bf16_t* __restrict__ wb, const bf16_t* __restrict__ obsb,
                    float* __restrict__ out, float* __restrict__ fs) {
  unsigned* cnt = (unsigned*)(fs + FS_BAR);
  unsigned* gen = cnt + 1;
  float*  hs  = fs + FS_HS;
  float*  fep = fs + FS_FEP;
  bf16_t* bss = (bf16_t*)(fs + FS_BF);
  bf16_t* t1b  = bss + BS_T1;
  bf16_t* updb = bss + BS_UPD;
  bf16_t* hsb  = bss + BS_HS;
  const int lane = threadIdx.x & 31;
  // transpose wave numbering so the 64 active waves of an H-GEMV cover all WGs
  const int wid  = ((int)(threadIdx.x >> 5)) * (int)gridDim.x + (int)blockIdx.x;
  const int flat = (int)blockIdx.x * BLK + (int)threadIdx.x;

  for (int t = 0; t < S; ++t) {
    const float* inp = obs + (size_t)t * H;      // fp32 view (prec / err)
    for (int l = 0; l < L; ++l) {
      const float* predl = out + (size_t)l * H;        // level-0 slot stays 0
      float*       encl  = out + (size_t)(L + l) * H;  // errs[-1] slot + next inp
      if (l > 0) {
        { // t1 = gelu(pw1[l] * hs[l-1] + pb1[l])
          const bf16_t* W = wb + (size_t)l * H * H;
          const float*  b = pb1 + (size_t)l * H;
          gemv_bf16<XBF16>(W, H, H, hsb + (size_t)(l - 1) * H, nullptr, wid,
                           [=](int m0, v8f a) {
#pragma unroll
                             for (int r = 0; r < 8; ++r)
                               t1b[m0 + r] = (bf16_t)gelu_exact(a[r] + b[m0 + r]);
                           });
          gbar(cnt, gen);
        }
        { // pred = pw2[l] * t1 + pb2[l]  (fp32, it is an output & err input)
          const bf16_t* W = wb + NHH + (size_t)l * H * H;
          const float*  b = pb2 + (size_t)l * H;
          float* po = out + (size_t)l * H;
          gemv_bf16<XBF16>(W, H, H, t1b, nullptr, wid,
                           [=](int m0, v8f a) {
#pragma unroll
                             for (int r = 0; r < 8; ++r) po[m0 + r] = a[r] + b[m0 + r];
                           });
          gbar(cnt, gen);
        }
      }
      { // t1 = gelu(ew1[l] * (inp - pred) + eb1[l]); prec on an idle wave
        const bf16_t* W = wb + 2 * NHH + (size_t)l * H * H;
        const float*  b = eb1 + (size_t)l * H;
        auto epi = [=](int m0, v8f a) {
#pragma unroll
          for (int r = 0; r < 8; ++r)
            t1b[m0 + r] = (bf16_t)gelu_exact(a[r] + b[m0 + r]);
        };
        if (l > 0)
          gemv_bf16<XF32SUB>(W, H, H, inp, predl, wid, epi);
        else  // err == observed: use the pre-converted bf16 mirror
          gemv_bf16<XBF16>(W, H, H, obsb + (size_t)t * H, nullptr, wid, epi);
        if (wid == 64) {  // prec = sigmoid(err . qw[l] + qb[l])
          const float* qwl = qw + (size_t)l * H;
          float s = 0.f;
          for (int i = lane; i < H; i += 32) {
            float e = inp[i] - (l > 0 ? predl[i] : 0.f);
            s += e * qwl[i];
          }
          for (int off = 16; off > 0; off >>= 1) s += __shfl_down(s, off, 32);
          if (lane == 0) fs[FS_PREC] = sigm(s + qb[l]);
        }
        gbar(cnt, gen);
      }
      { // enc = ew2[l]*t1 + eb2[l]; fused upd(bf16) + fe partials
        const bf16_t* W  = wb + 3 * NHH + (size_t)l * H * H;
        const float*  b  = eb2 + (size_t)l * H;
        const float* hsl = hs + (size_t)l * H;
        const float  pr  = fs[FS_PREC];
        float fes = 0.f;
        gemv_bf16<XBF16>(W, H, H, t1b, nullptr, wid,
                         [&](int m0, v8f a) {
#pragma unroll
                           for (int r = 0; r < 8; ++r) {
                             float e = a[r] + b[m0 + r];
                             encl[m0 + r] = e;
                             updb[m0 + r] = (bf16_t)(hsl[m0 + r] + e * pr);
                             fes += e * e;
                           }
                         });
        for (int off = 16; off > 0; off >>= 1) fes += __shfl_down(fes, off, 32);
        if (lane == 0) fep[wid] = fes;   // every wave writes (0 if idle)
        gbar(cnt, gen);
      }
      { // LSTM gates + activation fused (wih rows permuted to 4*j+gate):
        // each wave's 16-row tile holds all 4 gates of 4 outputs, so
        // h = sig(o)*tanh(sig(i)*tanh(g)) happens right in the epilogue.
        const bf16_t* W   = wb + 4 * NHH + (size_t)l * 4 * H * H;
        const float*  bi  = bih + (size_t)l * 4 * H;
        const float*  bh  = bhh + (size_t)l * 4 * H;
        float* hsl  = hs  + (size_t)l * H;
        bf16_t* hsbl = hsb + (size_t)l * H;
        gemv_bf16<XBF16>(W, 4 * H, H, updb, nullptr, wid,
                         [=](int m0, v8f a) {
                           const int j0 = m0 >> 2;   // m0 = 4*j0 (m0 % 8 == 0)
#pragma unroll
                           for (int jj = 0; jj < 2; ++jj) {
                             const int j = j0 + jj;
                             float gi = a[4 * jj + 0] + bi[j] + bh[j];
                             // f-gate (4*jj+1) is dead: c0 == 0
                             float gg = a[4 * jj + 2] + bi[2 * H + j] + bh[2 * H + j];
                             float go = a[4 * jj + 3] + bi[3 * H + j] + bh[3 * H + j];
                             float hv = sigm(go) * tanhf(sigm(gi) * tanhf(gg));
                             hsl[j] = hv; hsbl[j] = (bf16_t)hv;
                           }
                         });
        if (wid == 0 && lane == 0) {  // deterministic fixed-order fe reduction
          float s = 0.f;
          for (int i = 0; i < NWAVES; ++i) s += fep[i];
          fs[FS_FE] += fs[FS_PREC] * s;
        }
        gbar(cnt, gen);
      }
      inp = encl;  // next level consumes encoded error
    }
  }
  if (flat == 0) out[2 * L * H] = fs[FS_FE];  // out[8192]
}

extern "C" void kernel_launch(void* const* d_in, const int* in_sizes, int n_in,
                              void* d_out, int out_size, void* d_ws, size_t ws_size,
                              hipStream_t stream) {
  (void)in_sizes; (void)n_in; (void)out_size; (void)ws_size;
  const float* obs = (const float*)d_in[0];
  const float* pw1 = (const float*)d_in[1];
  const float* pb1 = (const float*)d_in[2];
  const float* pw2 = (const float*)d_in[3];
  const float* pb2 = (const float*)d_in[4];
  const float* ew1 = (const float*)d_in[5];
  const float* eb1 = (const float*)d_in[6];
  const float* ew2 = (const float*)d_in[7];
  const float* eb2 = (const float*)d_in[8];
  const float* qw  = (const float*)d_in[9];
  const float* qb  = (const float*)d_in[10];
  const float* wih = (const float*)d_in[11];
  /* d_in[12] = whh: unused (h0 == 0 each step, W_hh term vanishes) */
  const float* bih = (const float*)d_in[13];
  const float* bhh = (const float*)d_in[14];

  bf16_t* wb   = (bf16_t*)d_ws;                  // weights (64MB bf16)
  bf16_t* obsb = wb + WB_ELEMS;                  // observed mirror (1MB bf16)
  float*  fs   = (float*)((char*)d_ws + CONV_TOTAL * sizeof(bf16_t));
  float*  out  = (float*)d_out;

  convert_weights<<<2048, 256, 0, stream>>>(pw1, pw2, ew1, ew2, wih, obs, wb);
  init_state<<<4, 1024, 0, stream>>>(fs, out);
  tpn_persistent<<<NWG, BLK, 0, stream>>>(obs, pb1, pb2, eb1, eb2, qw, qb,
                                          bih, bhh, wb, obsb, out, fs);
}
